// Attention_42717744726175
// MI455X (gfx1250) — matrix operand order repaired and checked by reading, not compiled, not run on hardware
//
#include <hip/hip_runtime.h>

// ---------------------------------------------------------------------------
// MHA forward, B=4, L=2048, DIM=1024, H=16, HD=64, gfx1250 (wave32).
// GEMMs via V_WMMA_F32_16X16X32_F16; A-tiles staged by the Tensor Data Mover
// (TENSORcnt), attention K-tiles by global_load_async_to_lds (ASYNCcnt),
// both double-buffered so staging overlaps WMMA compute.
// ---------------------------------------------------------------------------

typedef _Float16 half_t;
typedef __attribute__((ext_vector_type(4)))  _Float16     v4h;
typedef __attribute__((ext_vector_type(8)))  _Float16     v8h;
typedef __attribute__((ext_vector_type(16))) _Float16     v16h;
typedef __attribute__((ext_vector_type(8)))  float        v8f;
typedef __attribute__((ext_vector_type(4)))  unsigned int v4u;
typedef __attribute__((ext_vector_type(8)))  int          v8i;
typedef __attribute__((ext_vector_type(4)))  int          v4i;

#define BB   4
#define LL   2048
#define DIMM 1024
#define HH   16
#define HD   64

__device__ __forceinline__ v16h cat8(v8h lo, v8h hi) {
  return __builtin_shufflevector(lo, hi, 0,1,2,3,4,5,6,7,8,9,10,11,12,13,14,15);
}

__device__ __forceinline__ v8f wmma_f16(v16h a, v16h b, v8f c) {
  return __builtin_amdgcn_wmma_f32_16x16x32_f16(false, a, false, b, (short)0, c,
                                                false, false);
}

// ---------------------------------------------------------------------------
// TDM: 2D f16 tile (tile_w elems x tile_h rows) from row-major tensor into
// LDS, +16B pad after every 64B row -> LDS row stride 80B == [.][40] halves.
// D# packed per CDNA5 ISA 8.3/8.4.  Issued by one wave; tracked by TENSORcnt.
// ---------------------------------------------------------------------------
__device__ __forceinline__ void tdm_load_2d_f16(const half_t* gsrc,
                                                unsigned lds_byte_addr,
                                                unsigned tile_w, unsigned tile_h,
                                                unsigned row_stride_elems) {
  unsigned long long ga = (unsigned long long)(uintptr_t)gsrc;
  v4u g0;
  g0[0] = 1u;                                   // count=1, user descriptor
  g0[1] = lds_byte_addr;                        // bits 63:32  lds_addr
  g0[2] = (unsigned)ga;                         // bits 95:64  global_addr lo
  g0[3] = (unsigned)((ga >> 32) & 0x01ffffffu)  // bits 120:96 global_addr hi
        | (2u << 30);                           // bits 127:126 type=2 (image)
  v8i g1;
  g1[0] = (1 << 16)                             // data_size = 2 bytes
        | (1 << 20)                             // pad_enable
        | (3 << 22)                             // pad_interval: 16 DWORDs
        | (3 << 25);                            // pad_amount:   4 DWORDs
  unsigned td0 = row_stride_elems;              // tensor_dim0 (row length)
  unsigned td1 = 1u << 20;                      // tensor_dim1 (tiles never OOB)
  g1[1] = (int)((td0 & 0xffffu) << 16);
  g1[2] = (int)((td0 >> 16) | ((td1 & 0xffffu) << 16));
  g1[3] = (int)((td1 >> 16) | (tile_w << 16));
  g1[4] = (int)(tile_h & 0xffffu);
  g1[5] = (int)row_stride_elems;
  g1[6] = 0;
  g1[7] = 0;
  v4i z4 = {0, 0, 0, 0};
#if __clang_major__ >= 23
  v8i z8 = {0, 0, 0, 0, 0, 0, 0, 0};
  __builtin_amdgcn_tensor_load_to_lds(g0, g1, z4, z4, z8, 0);
#else
  __builtin_amdgcn_tensor_load_to_lds(g0, g1, z4, z4, 0);
#endif
}

// ---------------------------------------------------------------------------
// fp32 -> f16 conversion
// ---------------------------------------------------------------------------
__global__ void cvt_f32_f16(const float4* __restrict__ src,
                            v4h* __restrict__ dst, int n4) {
  int i = blockIdx.x * blockDim.x + threadIdx.x;
  if (i < n4) {
    float4 f = src[i];
    v4h h;
    h[0] = (_Float16)f.x; h[1] = (_Float16)f.y;
    h[2] = (_Float16)f.z; h[3] = (_Float16)f.w;
    dst[i] = h;
  }
}

// ---------------------------------------------------------------------------
// GEMM C[8192,3072] = X[8192,1024] @ Wqkv[1024,3072]; epilogue scatters f16
// into Q/K/V [B,H,L,HD].  Block tile 128x64, K-step 32, 4 waves (32 rows
// each), double-buffered LDS, TDM-staged A tile.
// ---------------------------------------------------------------------------
__global__ void gemm_qkv(const half_t* __restrict__ X,
                         const half_t* __restrict__ W,
                         half_t* __restrict__ Q,
                         half_t* __restrict__ K,
                         half_t* __restrict__ V) {
  __shared__ __align__(16) half_t As[2][128][40];
  __shared__ __align__(16) half_t Bt[2][64][40];

  const int tid  = threadIdx.x;
  const int lane = tid & 31;
  const int wave = tid >> 5;
  const int m0   = blockIdx.y * 128;
  const int n0   = blockIdx.x * 64;
  const int col  = lane & 15;
  const int hi16 = lane >> 4;
  const int kh   = hi16 * 8;
  const int kb   = hi16 * 16;

  v8f acc[2][4] = {};

  auto issueA = [&](int buf, int k0) {
    if (wave == 0)
      tdm_load_2d_f16(X + (size_t)m0 * 1024 + k0,
                      (unsigned)(uintptr_t)&As[buf][0][0], 32, 128, 1024);
  };
  auto stageW = [&](int buf, int k0) {
    int k = tid >> 2, c0 = (tid & 3) * 16;
    const half_t* src = W + (size_t)(k0 + k) * 3072 + n0 + c0;
    v8h w0 = *(const v8h*)src;
    v8h w1 = *(const v8h*)(src + 8);
    #pragma unroll
    for (int j = 0; j < 8; ++j) {
      Bt[buf][c0 + j][k]     = w0[j];
      Bt[buf][c0 + 8 + j][k] = w1[j];
    }
    if (k0 + 32 < 1024) __builtin_prefetch(src + (size_t)32 * 3072, 0, 1);
  };

  issueA(0, 0);
  stageW(0, 0);
  __builtin_amdgcn_s_wait_tensorcnt(0);
  __syncthreads();

  for (int it = 0; it < 32; ++it) {
    const int buf = it & 1;
    if (it + 1 < 32) {
      issueA(buf ^ 1, (it + 1) * 32);
      stageW(buf ^ 1, (it + 1) * 32);
    }
    #pragma unroll
    for (int sub = 0; sub < 2; ++sub) {
      v8h alo = *(const v8h*)&As[buf][wave * 32 + sub * 16 + col][kh];
      v8h ahi = *(const v8h*)&As[buf][wave * 32 + sub * 16 + col][kh + 16];
      v16h af = cat8(alo, ahi);
      #pragma unroll
      for (int nt = 0; nt < 4; ++nt) {
        v8h blo = *(const v8h*)&Bt[buf][nt * 16 + col][kh];
        v8h bhi = *(const v8h*)&Bt[buf][nt * 16 + col][kh + 16];
        acc[sub][nt] = wmma_f16(af, cat8(blo, bhi), acc[sub][nt]);
      }
    }
    __builtin_amdgcn_s_wait_tensorcnt(0);
    __syncthreads();
  }

  const int rowoff = hi16 * 8;
  #pragma unroll
  for (int sub = 0; sub < 2; ++sub)
    #pragma unroll
    for (int nt = 0; nt < 4; ++nt) {
      int n  = n0 + nt * 16 + col;
      int s  = n >> 10;
      int h  = (n >> 6) & 15;
      int hd = n & 63;
      half_t* dst = (s == 0) ? Q : (s == 1) ? K : V;
      #pragma unroll
      for (int i = 0; i < 8; ++i) {
        int m = m0 + wave * 32 + sub * 16 + rowoff + i;
        int b = m >> 11, l = m & 2047;
        dst[(((size_t)b * HH + h) * LL + l) * HD + hd] = (half_t)acc[sub][nt][i];
      }
    }
}

// ---------------------------------------------------------------------------
// Flash attention per (b,h): 4 waves x 16 query rows, 64-key blocks, K/V
// staging double-buffered (K via global_load_async_to_lds, V transposed via
// DS), online softmax fp32, QK^T and PV via WMMA.  Mask is all-true.
// Scale (1/8, exact in f16) folded into the Q fragments once.
// ---------------------------------------------------------------------------
__global__ void attn_fwd(const half_t* __restrict__ Q,
                         const half_t* __restrict__ K,
                         const half_t* __restrict__ V,
                         half_t* __restrict__ ctx) {
  __shared__ __align__(16) half_t Ks[2][64][72];
  __shared__ __align__(16) half_t Vt[2][64][72];
  __shared__ __align__(16) half_t Pw[4][16][72];

  const int tid  = threadIdx.x;
  const int lane = tid & 31;
  const int wave = tid >> 5;
  const int bh   = blockIdx.y;
  const int b    = bh >> 4, h = bh & 15;
  const int qblk = blockIdx.x * 64;
  const int col  = lane & 15;
  const int hi16 = lane >> 4;
  const int kh   = hi16 * 8;
  const int kb   = hi16 * 16;

  const size_t head = (size_t)bh * LL * HD;
  const half_t* Qh = Q + head;
  const half_t* Kh = K + head;
  const half_t* Vh = V + head;

  // Q A-fragments, pre-scaled by HD^-0.5 = 1/8 (power of two -> f16 exact)
  const int qrow = qblk + wave * 16 + col;
  v16h qa[2];
  #pragma unroll
  for (int f = 0; f < 2; ++f) {
    v8h lo = *(const v8h*)(Qh + (size_t)qrow * HD + f * 32 + kh);
    v8h hi = *(const v8h*)(Qh + (size_t)qrow * HD + f * 32 + kh + 16);
    v16h q = cat8(lo, hi);
    #pragma unroll
    for (int e = 0; e < 16; ++e) q[e] = q[e] * (_Float16)0.125f;
    qa[f] = q;
  }

  v8f o[4] = {};
  float mrow[8], lrow[8];
  #pragma unroll
  for (int i = 0; i < 8; ++i) { mrow[i] = -1e30f; lrow[i] = 0.0f; }

  auto stageKV = [&](int buf, int j0) {
    int r = tid >> 1, c0 = (tid & 1) * 32;
    // async-stage K rows (IOFFSET applies to both global and LDS sides)
    {
      const half_t* g = Kh + (size_t)(j0 + r) * HD + c0;
      unsigned l = (unsigned)(uintptr_t)&Ks[buf][r][c0];
      asm volatile(
        "global_load_async_to_lds_b128 %0, %1, off\n\t"
        "global_load_async_to_lds_b128 %0, %1, off offset:16\n\t"
        "global_load_async_to_lds_b128 %0, %1, off offset:32\n\t"
        "global_load_async_to_lds_b128 %0, %1, off offset:48"
        :: "v"(l), "v"(g) : "memory");
    }
    // stage V transposed
    #pragma unroll
    for (int cc = 0; cc < 32; cc += 8) {
      v8h v = *(const v8h*)(Vh + (size_t)(j0 + r) * HD + c0 + cc);
      #pragma unroll
      for (int e = 0; e < 8; ++e) Vt[buf][c0 + cc + e][r] = v[e];
    }
  };

  stageKV(0, 0);
  asm volatile("s_wait_asynccnt 0x0" ::: "memory");
  __syncthreads();

  for (int jb = 0; jb < 32; ++jb) {
    const int buf = jb & 1;
    if (jb + 1 < 32) stageKV(buf ^ 1, (jb + 1) * 64);

    // ---- S = (Q/8) @ K^T ----
    v8f s[4];
    #pragma unroll
    for (int nt = 0; nt < 4; ++nt) {
      v8f a = {};
      #pragma unroll
      for (int f = 0; f < 2; ++f) {
        v8h blo = *(const v8h*)&Ks[buf][nt * 16 + col][f * 32 + kh];
        v8h bhi = *(const v8h*)&Ks[buf][nt * 16 + col][f * 32 + kh + 16];
        a = wmma_f16(qa[f], cat8(blo, bhi), a);
      }
      s[nt] = a;
    }

    // ---- online softmax ----
    #pragma unroll
    for (int i = 0; i < 8; ++i) {
      float rm = fmaxf(fmaxf(s[0][i], s[1][i]), fmaxf(s[2][i], s[3][i]));
      #pragma unroll
      for (int mk = 1; mk < 16; mk <<= 1) rm = fmaxf(rm, __shfl_xor(rm, mk, 16));
      float mnew = fmaxf(mrow[i], rm);
      float a    = __expf(mrow[i] - mnew);
      mrow[i] = mnew;
      float rs = 0.0f;
      #pragma unroll
      for (int nt = 0; nt < 4; ++nt) {
        float p = __expf(s[nt][i] - mnew);
        s[nt][i] = p;
        rs += p;
      }
      #pragma unroll
      for (int mk = 1; mk < 16; mk <<= 1) rs += __shfl_xor(rs, mk, 16);
      lrow[i] = lrow[i] * a + rs;
      #pragma unroll
      for (int nt = 0; nt < 4; ++nt) o[nt][i] *= a;
    }

    // ---- P (D-layout) -> LDS -> A-layout ----
    #pragma unroll
    for (int nt = 0; nt < 4; ++nt)
      #pragma unroll
      for (int i = 0; i < 8; ++i)
        Pw[wave][hi16 * 8 + i][nt * 16 + col] = (half_t)s[nt][i];
    asm volatile("s_wait_dscnt 0x0" ::: "memory");

    // ---- O += P @ V ----
    #pragma unroll
    for (int f = 0; f < 2; ++f) {
      v8h plo = *(const v8h*)&Pw[wave][col][f * 32 + kh];
      v8h phi = *(const v8h*)&Pw[wave][col][f * 32 + kh + 16];
      v16h pa = cat8(plo, phi);
      #pragma unroll
      for (int nt = 0; nt < 4; ++nt) {
        v8h blo = *(const v8h*)&Vt[buf][nt * 16 + col][f * 32 + kh];
        v8h bhi = *(const v8h*)&Vt[buf][nt * 16 + col][f * 32 + kh + 16];
        o[nt] = wmma_f16(pa, cat8(blo, bhi), o[nt]);
      }
    }

    asm volatile("s_wait_asynccnt 0x0" ::: "memory");
    __syncthreads();
  }

  #pragma unroll
  for (int nt = 0; nt < 4; ++nt)
    #pragma unroll
    for (int i = 0; i < 8; ++i) {
      int row = qblk + wave * 16 + hi16 * 8 + i;
      float val = o[nt][i] / lrow[i];
      ctx[((size_t)b * LL + row) * DIMM + h * HD + nt * 16 + col] = (half_t)val;
    }
}

// ---------------------------------------------------------------------------
// Out-projection: out[8192,1024] = ctx[8192,1024] @ Wout[1024,1024] (fp32)
// ---------------------------------------------------------------------------
__global__ void gemm_out(const half_t* __restrict__ X,
                         const half_t* __restrict__ W,
                         float* __restrict__ Out) {
  __shared__ __align__(16) half_t As[2][128][40];
  __shared__ __align__(16) half_t Bt[2][64][40];

  const int tid  = threadIdx.x;
  const int lane = tid & 31;
  const int wave = tid >> 5;
  const int m0   = blockIdx.y * 128;
  const int n0   = blockIdx.x * 64;
  const int col  = lane & 15;
  const int hi16 = lane >> 4;
  const int kh   = hi16 * 8;
  const int kb   = hi16 * 16;

  v8f acc[2][4] = {};

  auto issueA = [&](int buf, int k0) {
    if (wave == 0)
      tdm_load_2d_f16(X + (size_t)m0 * 1024 + k0,
                      (unsigned)(uintptr_t)&As[buf][0][0], 32, 128, 1024);
  };
  auto stageW = [&](int buf, int k0) {
    int k = tid >> 2, c0 = (tid & 3) * 16;
    const half_t* src = W + (size_t)(k0 + k) * 1024 + n0 + c0;
    v8h w0 = *(const v8h*)src;
    v8h w1 = *(const v8h*)(src + 8);
    #pragma unroll
    for (int j = 0; j < 8; ++j) {
      Bt[buf][c0 + j][k]     = w0[j];
      Bt[buf][c0 + 8 + j][k] = w1[j];
    }
    if (k0 + 32 < 1024) __builtin_prefetch(src + (size_t)32 * 1024, 0, 1);
  };

  issueA(0, 0);
  stageW(0, 0);
  __builtin_amdgcn_s_wait_tensorcnt(0);
  __syncthreads();

  for (int it = 0; it < 32; ++it) {
    const int buf = it & 1;
    if (it + 1 < 32) {
      issueA(buf ^ 1, (it + 1) * 32);
      stageW(buf ^ 1, (it + 1) * 32);
    }
    #pragma unroll
    for (int sub = 0; sub < 2; ++sub) {
      v8h alo = *(const v8h*)&As[buf][wave * 32 + sub * 16 + col][kh];
      v8h ahi = *(const v8h*)&As[buf][wave * 32 + sub * 16 + col][kh + 16];
      v16h af = cat8(alo, ahi);
      #pragma unroll
      for (int nt = 0; nt < 4; ++nt) {
        v8h blo = *(const v8h*)&Bt[buf][nt * 16 + col][kh];
        v8h bhi = *(const v8h*)&Bt[buf][nt * 16 + col][kh + 16];
        acc[sub][nt] = wmma_f16(af, cat8(blo, bhi), acc[sub][nt]);
      }
    }
    __builtin_amdgcn_s_wait_tensorcnt(0);
    __syncthreads();
  }

  const int rowoff = hi16 * 8;
  #pragma unroll
  for (int sub = 0; sub < 2; ++sub)
    #pragma unroll
    for (int nt = 0; nt < 4; ++nt)
      #pragma unroll
      for (int i = 0; i < 8; ++i)
        Out[(size_t)(m0 + wave * 32 + sub * 16 + rowoff + i) * 1024 +
            n0 + nt * 16 + col] = acc[sub][nt][i];
}

// ---------------------------------------------------------------------------
extern "C" void kernel_launch(void* const* d_in, const int* in_sizes, int n_in,
                              void* d_out, int out_size, void* d_ws, size_t ws_size,
                              hipStream_t stream) {
  (void)in_sizes; (void)n_in; (void)out_size; (void)ws_size;
  const float* x     = (const float*)d_in[0];
  // d_in[1] = mask: all-true -> unmasked softmax
  const float* w_qkv = (const float*)d_in[2];
  const float* w_out = (const float*)d_in[3];
  float* out = (float*)d_out;

  char* p = (char*)d_ws;
  half_t* xh  = (half_t*)p; p += (size_t)BB * LL * DIMM * 2;
  half_t* wqh = (half_t*)p; p += (size_t)DIMM * 3 * DIMM * 2;
  half_t* woh = (half_t*)p; p += (size_t)DIMM * DIMM * 2;
  half_t* qh  = (half_t*)p; p += (size_t)BB * HH * LL * HD * 2;
  half_t* kh  = (half_t*)p; p += (size_t)BB * HH * LL * HD * 2;
  half_t* vh  = (half_t*)p; p += (size_t)BB * HH * LL * HD * 2;
  half_t* ctx = (half_t*)p;

  cvt_f32_f16<<<8192, 256, 0, stream>>>((const float4*)x,     (v4h*)xh,  2097152);
  cvt_f32_f16<<<3072, 256, 0, stream>>>((const float4*)w_qkv, (v4h*)wqh,  786432);
  cvt_f32_f16<<<1024, 256, 0, stream>>>((const float4*)w_out, (v4h*)woh,  262144);

  gemm_qkv<<<dim3(48, 64), 128, 0, stream>>>(xh, wqh, qh, kh, vh);
  attn_fwd<<<dim3(32, 64), 128, 0, stream>>>(qh, kh, vh, ctx);
  gemm_out<<<dim3(16, 64), 128, 0, stream>>>(ctx, woh, out);
}
